// GRU_34995393528140
// MI455X (gfx1250) — matrix operand order repaired and checked by reading, not compiled, not run on hardware
//
#include <hip/hip_runtime.h>

#define VOCAB 32000
#define EMB   256
#define HID   512
#define BB    16
#define SS    256
#define MTOT  (BB*SS)      /* 4096 */
#define N3H   (3*HID)      /* 1536 */
#define SCAN_WGS 8

typedef __attribute__((ext_vector_type(16))) __bf16 v16bf;
typedef __attribute__((ext_vector_type(8)))  float  v8f;
typedef __attribute__((ext_vector_type(4)))  unsigned int v4u;
typedef __attribute__((ext_vector_type(8)))  int v8i;
typedef __attribute__((ext_vector_type(4)))  int v4i;

union Frag {
  uint4          q[2];
  unsigned short s[16];
  v16bf          v;
};

__device__ __forceinline__ unsigned short f2bf(float f) {
  unsigned u = __float_as_uint(f);
  u += 0x7FFFu + ((u >> 16) & 1u);   // round-to-nearest-even
  return (unsigned short)(u >> 16);
}

__device__ __forceinline__ v8f wmma_bf16(const Frag& a, const Frag& b, v8f c) {
  return __builtin_amdgcn_wmma_f32_16x16x32_bf16(false, a.v, false, b.v,
                                                 (short)0, c, false, false);
}

// ---------------------------------------------------------------------------
// K0a: f32 [K][N] -> bf16 transposed [N][K]
// ---------------------------------------------------------------------------
__global__ void transpose_to_bf16(const float* __restrict__ in,
                                  unsigned short* __restrict__ out,
                                  int K, int N) {
  int total = K * N;
  for (int i = blockIdx.x * blockDim.x + threadIdx.x; i < total;
       i += gridDim.x * blockDim.x) {
    int n = i / K;
    int k = i - n * K;
    out[i] = f2bf(in[k * N + n]);
  }
}

// ---------------------------------------------------------------------------
// K0b: zero h buffers (double-buffered) + barrier state
// ---------------------------------------------------------------------------
__global__ void init_state(float* __restrict__ h_f32,
                           unsigned short* __restrict__ h_bf16,
                           unsigned* __restrict__ bar) {
  int i = blockIdx.x * blockDim.x + threadIdx.x;
  if (i < 2 * BB * HID) { h_f32[i] = 0.0f; h_bf16[i] = 0; }
  if (i < 2) bar[i] = 0u;
}

// ---------------------------------------------------------------------------
// K1: xw[m][n] = E[x[m]] . W[:,n] + b0[n]   (M=4096, K=256, N=1536)
// grid = (12, 256), 256 thr (8 waves); wave tile 16x16, WG tile 16x128.
// ---------------------------------------------------------------------------
__global__ void __launch_bounds__(256)
embed_xw_kernel(const int* __restrict__ x, const float* __restrict__ E,
                const unsigned short* __restrict__ WT,
                const float* __restrict__ bias, float* __restrict__ xw) {
  const int lane = threadIdx.x & 31;
  const int wave = threadIdx.x >> 5;            // 0..7
  const int m0   = blockIdx.y * 16;
  const int n0   = blockIdx.x * 128 + wave * 16;
  const int rr   = lane & 15;
  const bool hi  = lane >= 16;

  const int   m    = m0 + rr;                   // A row handled by this lane
  const int   idx  = x[m];                      // embedding gather
  const float* erow = E + (long)idx * EMB;
  const unsigned short* brow = WT + (n0 + rr) * EMB;

  v8f acc = {};
#pragma unroll
  for (int k0 = 0; k0 < EMB; k0 += 32) {
    const int ka = k0 + (hi ? 8 : 0);
    union { float4 f4[4]; float f[16]; } t;
    t.f4[0] = *(const float4*)(erow + ka);
    t.f4[1] = *(const float4*)(erow + ka + 4);
    t.f4[2] = *(const float4*)(erow + ka + 16);
    t.f4[3] = *(const float4*)(erow + ka + 20);
    Frag a;
#pragma unroll
    for (int i = 0; i < 16; ++i) a.s[i] = f2bf(t.f[i]);
    const int kb = k0 + (hi ? 8 : 0);
    Frag b;
    b.q[0] = *(const uint4*)(brow + kb);
    b.q[1] = *(const uint4*)(brow + kb + 16);
    acc = wmma_bf16(a, b, acc);
  }
  const int   mb = m0 + (hi ? 8 : 0);
  const int   n  = n0 + rr;
  const float bv = bias[n];
#pragma unroll
  for (int r = 0; r < 8; ++r) xw[(mb + r) * N3H + n] = acc[r] + bv;
}

// ---------------------------------------------------------------------------
// device-wide barrier via L2 atomics (8 resident WGs, one barrier per step)
// ---------------------------------------------------------------------------
__device__ __forceinline__ void grid_sync(unsigned* cnt, unsigned* gen,
                                          unsigned target) {
  __syncthreads();
  if (threadIdx.x == 0) {
    __threadfence();
    unsigned arr = atomicAdd(cnt, 1u);
    if (arr == SCAN_WGS - 1u) {
      atomicExch(cnt, 0u);
      __threadfence();
      atomicExch(gen, target);
    } else {
      while (atomicAdd(gen, 0u) < target) { __builtin_amdgcn_s_sleep(2); }
      __threadfence();
    }
  }
  __syncthreads();
}

// ---------------------------------------------------------------------------
// K2: GRU scan. 8 WGs x 384 thr (12 waves). WG w owns h columns [64w,64w+64).
// ---------------------------------------------------------------------------
__global__ void __launch_bounds__(384)
gru_scan_kernel(const float* __restrict__ xw,
                const unsigned short* __restrict__ UT,   // [1536][512] bf16
                const float* __restrict__ bias,          // b[0],b[1] flat
                float* __restrict__ h_f32,               // [2][16][512]
                unsigned short* __restrict__ h_bf16,     // [2][16][512]
                unsigned short* __restrict__ h_all,      // [4096][512] bf16
                unsigned* __restrict__ bar) {
  const int wg   = blockIdx.x;                 // 0..7
  const int tid  = threadIdx.x;
  const int lane = tid & 31;
  const int wave = tid >> 5;                   // 0..11
  const int g    = wave >> 2;                  // gate 0..2
  const int ct   = wave & 3;                   // col tile 0..3
  const int col0 = wg * 64 + ct * 16;          // h-column base of wave tile
  const int n0   = g * HID + col0;             // rec column base
  const int rr   = lane & 15;
  const bool hi  = lane >= 16;

  __shared__ float rec[3][BB][64];

  const unsigned short* brow = UT + (n0 + rr) * HID;

  for (int s = 0; s < SS; ++s) {
    const int ph = s & 1;
    const unsigned short* hb = h_bf16 + ph * (BB * HID);

    // prefetch this step's gate inputs while the recurrent GEMM runs
    {
      const int bi = tid >> 6;
      const int ci = tid & 63;
      const int m  = bi * SS + s;
      __builtin_prefetch(&xw[m * N3H + wg * 64 + ci], 0, 0);
      __builtin_prefetch(&xw[m * N3H + HID + wg * 64 + ci], 0, 0);
      __builtin_prefetch(&xw[m * N3H + 2 * HID + wg * 64 + ci], 0, 0);
    }

    // ---- rec tile = h(16x512,bf16) x U-tile(512x16) ----
    v8f acc = {};
    const unsigned short* arow = hb + rr * HID;
    for (int k0 = 0; k0 < HID; k0 += 32) {
      Frag a, b;
      const int ka = k0 + (hi ? 8 : 0);
      a.q[0] = *(const uint4*)(arow + ka);
      a.q[1] = *(const uint4*)(arow + ka + 16);
      const int kb = k0 + (hi ? 8 : 0);
      b.q[0] = *(const uint4*)(brow + kb);
      b.q[1] = *(const uint4*)(brow + kb + 16);
      acc = wmma_bf16(a, b, acc);
    }
    {
      const int c  = ct * 16 + rr;
      const int mb = hi ? 8 : 0;
#pragma unroll
      for (int r = 0; r < 8; ++r) rec[g][mb + r][c] = acc[r];
    }
    __syncthreads();

    // ---- gates for this WG's 16x64 slice ----
    for (int e = tid; e < BB * 64; e += 384) {
      const int bi = e >> 6;                   // batch row
      const int ci = e & 63;                   // local col
      const int gc = wg * 64 + ci;             // global h col
      const int m  = bi * SS + s;              // flat (b,s)
      const float xz = xw[m * N3H + gc];
      const float xr = xw[m * N3H + HID + gc];
      const float xh = xw[m * N3H + 2 * HID + gc];
      const float rz = rec[0][bi][ci] + bias[N3H + gc];
      const float rr2 = rec[1][bi][ci] + bias[N3H + HID + gc];
      const float rh = rec[2][bi][ci] + bias[N3H + 2 * HID + gc];
      const float z  = 1.0f / (1.0f + __expf(-(xz + rz)));
      const float rg = 1.0f / (1.0f + __expf(-(xr + rr2)));
      const float hh = tanhf(xh + rg * rh);
      const float ho = h_f32[ph * (BB * HID) + bi * HID + gc];
      const float hn = z * ho + (1.0f - z) * hh;
      h_f32[(ph ^ 1) * (BB * HID) + bi * HID + gc] = hn;
      const unsigned short hb16 = f2bf(hn);
      h_bf16[(ph ^ 1) * (BB * HID) + bi * HID + gc] = hb16;
      h_all[(long)m * HID + gc] = hb16;
    }
    grid_sync(bar, bar + 1, (unsigned)(s + 1));
  }
}

// ---------------------------------------------------------------------------
// K3: logits = h_all(4096x512,bf16) x WdT^T + bd   (N=32000)
// grid=(250,64), 256 thr (8 waves). WG tile 64x128; wave tile 32x32.
// A block (64x512 bf16) staged into LDS by the Tensor Data Mover with row
// padding (4 DWORDs per 1024B row -> 1040B stride, conflict-free ds reads).
// ---------------------------------------------------------------------------
#define AROW 520   /* padded row stride in bf16 elements: 512 + 8 */

__global__ void __launch_bounds__(256)
decoder_kernel(const unsigned short* __restrict__ A,    // [4096][512] bf16
               const unsigned short* __restrict__ Bt,   // [32000][512] bf16
               const float* __restrict__ bd, float* __restrict__ out) {
  __shared__ unsigned short Atile[64 * AROW];  // only LDS object -> offset 0

  const int lane = threadIdx.x & 31;
  const int wave = threadIdx.x >> 5;           // 0..7
  const int mi   = (wave >> 2) * 32;           // 0 or 32 within WG M-block
  const int ni   = (wave & 3) * 32;            // 0,32,64,96 within WG N-block
  const int m0   = blockIdx.y * 64;            // global M base of WG
  const int n0   = blockIdx.x * 128 + ni;      // global N base of wave
  const int rr   = lane & 15;
  const bool hi  = lane >= 16;

  // ---- TDM: stage A[m0 .. m0+64) x [0..512) into LDS (wave 0 issues) ----
  if (wave == 0) {
    unsigned long long ga = (unsigned long long)(A + (size_t)m0 * HID);
    v4u g0;
    g0.x = 1u;                                   // count=1, user descriptor
    g0.y = 0u;                                   // lds_addr = 0 (Atile)
    g0.z = (unsigned)(ga & 0xFFFFFFFFu);         // global_addr[31:0]
    g0.w = (unsigned)((ga >> 32) & 0x1FFFFFFu) | (2u << 30);  // [56:32]|type=2
    v8i g1;
    g1[0] = (int)((1u << 16)        // data_size = 2 bytes
                | (1u << 20)        // pad_enable
                | (7u << 22)        // pad_interval: every 256 DWORDs (1024B)
                | (3u << 25));      // pad_amount: 4 DWORDs (16B)
    g1[1] = (int)(512u << 16);      // tensor_dim0 = 512 (low 16)
    g1[2] = (int)(4096u << 16);     // tensor_dim0 hi=0 | tensor_dim1 = 4096
    g1[3] = (int)(512u << 16);      // tensor_dim1 hi=0 | tile_dim0 = 512
    g1[4] = 64;                     // tile_dim1 = 64, tile_dim2 = 0
    g1[5] = 512;                    // tensor_dim0_stride = 512 (low 32)
    g1[6] = 0;
    g1[7] = 0;
    v4i z4 = {0, 0, 0, 0};
#if __clang_major__ >= 23
    v8i z8 = {0, 0, 0, 0, 0, 0, 0, 0};
    __builtin_amdgcn_tensor_load_to_lds(g0, g1, z4, z4, z8, 0);
#else
    __builtin_amdgcn_tensor_load_to_lds(g0, g1, z4, z4, 0);
#endif
    __builtin_amdgcn_s_wait_tensorcnt(0);
  }
  // Escape Atile into the asm + memory clobber: the TDM write to LDS is not
  // visible to the compiler (descriptor carries the address as an integer),
  // so without this the ds loads below would fold to undef.
  asm volatile("" :: "s"(&Atile[0]) : "memory");
  __syncthreads();

  const unsigned short* brow0 = Bt + (long)(n0 + rr) * HID;
  const unsigned short* brow1 = Bt + (long)(n0 + 16 + rr) * HID;
  const unsigned short* arow0 = Atile + (mi + rr) * AROW;
  const unsigned short* arow1 = Atile + (mi + 16 + rr) * AROW;

  v8f acc00 = {}, acc01 = {}, acc10 = {}, acc11 = {};
#pragma unroll 4
  for (int k0 = 0; k0 < HID; k0 += 32) {
    Frag a0, a1, b0, b1;
    const int ka = k0 + (hi ? 8 : 0);
    a0.q[0] = *(const uint4*)(arow0 + ka);
    a0.q[1] = *(const uint4*)(arow0 + ka + 16);
    a1.q[0] = *(const uint4*)(arow1 + ka);
    a1.q[1] = *(const uint4*)(arow1 + ka + 16);
    const int kb = k0 + (hi ? 8 : 0);
    b0.q[0] = *(const uint4*)(brow0 + kb);
    b0.q[1] = *(const uint4*)(brow0 + kb + 16);
    b1.q[0] = *(const uint4*)(brow1 + kb);
    b1.q[1] = *(const uint4*)(brow1 + kb + 16);
    acc00 = wmma_bf16(a0, b0, acc00);
    acc01 = wmma_bf16(a0, b1, acc01);
    acc10 = wmma_bf16(a1, b0, acc10);
    acc11 = wmma_bf16(a1, b1, acc11);
  }

  const int   gm0 = m0 + mi + (hi ? 8 : 0);
  const int   c0  = n0 + rr;
  const int   c1  = n0 + 16 + rr;
  const float bv0 = bd[c0];
  const float bv1 = bd[c1];
#pragma unroll
  for (int r = 0; r < 8; ++r) {
    out[(long)(gm0 + r) * VOCAB + c0]      = acc00[r] + bv0;
    out[(long)(gm0 + r) * VOCAB + c1]      = acc01[r] + bv1;
    out[(long)(gm0 + 16 + r) * VOCAB + c0] = acc10[r] + bv0;
    out[(long)(gm0 + 16 + r) * VOCAB + c1] = acc11[r] + bv1;
  }
}

// ---------------------------------------------------------------------------
extern "C" void kernel_launch(void* const* d_in, const int* in_sizes, int n_in,
                              void* d_out, int out_size, void* d_ws,
                              size_t ws_size, hipStream_t stream) {
  const int*   x    = (const int*)d_in[0];
  const float* E    = (const float*)d_in[1];
  const float* W    = (const float*)d_in[2];
  const float* U    = (const float*)d_in[3];
  const float* bias = (const float*)d_in[4];   // (2,1536) flat
  const float* Wd   = (const float*)d_in[5];
  const float* bd   = (const float*)d_in[6];
  float* out = (float*)d_out;

  char* ws = (char*)d_ws;
  size_t off = 0;
  unsigned short* WT  = (unsigned short*)(ws + off); off += (size_t)N3H * EMB * 2;      // 768 KB
  unsigned short* UT  = (unsigned short*)(ws + off); off += (size_t)N3H * HID * 2;      // 1.5 MB
  unsigned short* WdT = (unsigned short*)(ws + off); off += (size_t)VOCAB * HID * 2;    // 32 MB
  float*          xw  = (float*)(ws + off);          off += (size_t)MTOT * N3H * 4;     // 24 MB
  float*          hf  = (float*)(ws + off);          off += (size_t)2 * BB * HID * 4;
  unsigned short* hb  = (unsigned short*)(ws + off); off += (size_t)2 * BB * HID * 2;
  unsigned short* hA  = (unsigned short*)(ws + off); off += (size_t)MTOT * HID * 2;     // 4 MB
  unsigned*       bar = (unsigned*)(ws + off);       off += 64;

  transpose_to_bf16<<<512, 256, 0, stream>>>(W, WT, EMB, N3H);
  transpose_to_bf16<<<1024, 256, 0, stream>>>(U, UT, HID, N3H);
  transpose_to_bf16<<<4096, 256, 0, stream>>>(Wd, WdT, HID, VOCAB);
  init_state<<<(2 * BB * HID + 255) / 256, 256, 0, stream>>>(hf, hb, bar);

  embed_xw_kernel<<<dim3(N3H / 128, MTOT / 16), 256, 0, stream>>>(x, E, WT, bias, xw);
  gru_scan_kernel<<<SCAN_WGS, 384, 0, stream>>>(xw, UT, bias, hf, hb, hA, bar);
  decoder_kernel<<<dim3(VOCAB / 128, MTOT / 64), 256, 0, stream>>>(hA, WdT, bd, out);
}
